// BktModel_26852135535054
// MI455X (gfx1250) — matrix operand order repaired and checked by reading, not compiled, not run on hardware
//
#include <hip/hip_runtime.h>
#include <hip/hip_bf16.h>

typedef __attribute__((ext_vector_type(16))) _Float16 v16h;
typedef __attribute__((ext_vector_type(8)))  float    v8f;

#define TLEN   1024
#define NCHUNK 128

static __device__ __forceinline__ v16h splat16(_Float16 x) {
  v16h v;
#pragma unroll
  for (int i = 0; i < 16; ++i) v[i] = x;
  return v;
}

static __device__ __forceinline__ float sigf(float x) {
  return 1.0f / (1.0f + __expf(-x));
}

// One wave32 handles 16 batch rows for all T.
// A-matrix (16x256 f16) = trajectory weights W, 8 fragments of 16x32.
// B-matrix (256x16 f16, constant): col n<8 = traj[:,n] mask (bit 7-n of j),
// col 8 = ones, cols 9..15 = 0.  D[:,k] = masked sums, D[:,8] = total.
__global__ __launch_bounds__(32)
void fastbkt_wmma_kernel(const float* __restrict__ corr,
                         const int*   __restrict__ kc,
                         const float* __restrict__ logits_table,
                         float*       __restrict__ out)
{
  const int lane   = threadIdx.x & 31;
  const int laneHi = lane >> 4;
  const int bl     = lane & 15;
  const int b      = blockIdx.x * 16 + bl;

  __shared__ float cst[16][8];   // Ec1,Ec0,Ei1,Ei0,pF1,pL1,pF0,pL0 per row
  __shared__ float nbuf[16][2];  // normalized hidden-state mass (N1,N0)

  // ---- per-row probability-domain constants (exp(log_sigmoid) = sigmoid) ----
  const int kcb = kc[b];
  const float lpL = logits_table[kcb*5+0];
  const float lpF = logits_table[kcb*5+1];
  const float lpG = logits_table[kcb*5+2];
  const float lpS = logits_table[kcb*5+3];
  const float lpI = logits_table[kcb*5+4];

  const float pL1 = sigf(lpL),  pL0 = sigf(-lpL);
  const float pF0 = sigf(lpF),  pF1 = sigf(-lpF);
  const float Ec1 = sigf(-lpS), Ec0 = sigf(lpG);
  const float Ei1 = sigf(lpS),  Ei0 = sigf(-lpG);
  const float PI1 = sigf(lpI),  PI0 = sigf(-lpI);

  if (lane < 16) {
    cst[bl][0]=Ec1; cst[bl][1]=Ec0; cst[bl][2]=Ei1; cst[bl][3]=Ei0;
    cst[bl][4]=pF1; cst[bl][5]=pL1; cst[bl][6]=pF0; cst[bl][7]=pL0;
    nbuf[bl][0]=PI1; nbuf[bl][1]=PI0;     // chunk-0 initial distribution
  }

  // ---- exp(base[j]) in A-fragment layout: j = 32q + ((s<8)?s:s+8) + 8*laneHi
  v16h EB[8];
#pragma unroll
  for (int q = 0; q < 8; ++q) {
#pragma unroll
    for (int s = 0; s < 16; ++s) {
      const int j = 32*q + ((s < 8) ? s : s + 8) + 8*laneHi;
      float p = 1.0f;
#pragma unroll
      for (int k = 1; k < 8; ++k) {           // 7 Markov transitions
        const int prev = (j >> (8-k)) & 1;
        const int cur  = (j >> (7-k)) & 1;
        p *= prev ? (cur ? pF1 : pF0) : (cur ? pL1 : pL0);
      }
      EB[q][s] = (_Float16)p;
    }
  }

  // ---- constant B fragments: K row j = 32q + ((s<8)?s:s+8) + 8*laneHi, col n = lane&15
  v16h Bf[8];
#pragma unroll
  for (int q = 0; q < 8; ++q) {
#pragma unroll
    for (int s = 0; s < 16; ++s) {
      const int j = 32*q + ((s < 8) ? s : s + 8) + 8*laneHi;
      float v;
      if (bl < 8)       v = (float)((j >> (7-bl)) & 1);  // traj[:,bl] mask
      else if (bl == 8) v = 1.0f;                        // ones column
      else              v = 0.0f;
      Bf[q][s] = (_Float16)v;
    }
  }

  __syncthreads();

  const float* crow  = corr + (size_t)b * TLEN;
  const int   srcTot = (lane & 16) | 8;   // lane holding the "ones" column
  v16h W[8];

  for (int c = 0; c < NCHUNK; ++c) {
    if (c + 1 < NCHUNK) __builtin_prefetch(crow + (c+1)*8, 0, 0);

    // W init: W[j] = expBase[j] * (traj0(j) ? N1 : N0); traj0 = bit7 = (q&4)
    const _Float16 h1i = (_Float16)nbuf[bl][0];
    const _Float16 h0i = (_Float16)nbuf[bl][1];
#pragma unroll
    for (int q = 0; q < 8; ++q)
      W[q] = EB[q] * splat16((q & 4) ? h1i : h0i);

    // this chunk's observations (corr in {0,1} as f32)
    const float4 c4a = *(const float4*)(crow + c*8);
    const float4 c4b = *(const float4*)(crow + c*8 + 4);
    const float cobs[8] = {c4a.x,c4a.y,c4a.z,c4a.w,c4b.x,c4b.y,c4b.z,c4b.w};

#pragma unroll
    for (int k = 0; k < 8; ++k) {
      // ---- masked reductions over 256 trajectories (K = 8 x 32) ----
      v8f d = {};
#pragma unroll
      for (int q = 0; q < 8; ++q)
        d = __builtin_amdgcn_wmma_f32_16x16x32_f16(false, W[q], false, Bf[q],
                                                   (short)0, d, false, false);
      float s1[8], stot[8];
#pragma unroll
      for (int r = 0; r < 8; ++r) {
        s1[r]   = d[r];
        stot[r] = __shfl(d[r], srcTot, 32);   // broadcast total column
      }
      if (bl == k) {                          // lanes k, k+16 own column k
        const int t = c*8 + k;
#pragma unroll
        for (int r = 0; r < 8; ++r) {
          const int   m  = r + 8*laneHi;      // D rows: M = r (+8 hi half)
          const float S1 = s1[r];
          const float S0 = stot[r] - S1;
          const float Pc = cst[m][0]*S1 + cst[m][1]*S0;
          const float Pi = cst[m][2]*S1 + cst[m][3]*S0;
          const float iv = __frcp_rn(Pc + Pi);
          float2 o;                           // [pi, pc] already normalized
          o.x = __logf(Pi * iv);
          o.y = __logf(Pc * iv);
          *(float2*)(out + (((size_t)(blockIdx.x*16 + m))*TLEN + t)*2) = o;
        }
      }
      // ---- W *= obs factor; traj bit (7-k) maps to fragment/slot/lane bits
      const bool     cb = cobs[k] > 0.5f;
      const _Float16 h1 = (_Float16)(cb ? Ec1 : Ei1);
      const _Float16 h0 = (_Float16)(cb ? Ec0 : Ei0);
      if (k <= 2) {                           // bit in fragment index q
        const int qb = (k == 0) ? 4 : (k == 1) ? 2 : 1;
#pragma unroll
        for (int q = 0; q < 8; ++q) W[q] = W[q] * splat16((q & qb) ? h1 : h0);
      } else if (k == 4) {                    // bit = lane half
        const v16h fv = splat16(laneHi ? h1 : h0);
#pragma unroll
        for (int q = 0; q < 8; ++q) W[q] = W[q] * fv;
      } else {                                // bit in slot index (compile-time)
        const int bm = (k == 3) ? 8 : (k == 5) ? 4 : (k == 6) ? 2 : 1;
        v16h fv;
#pragma unroll
        for (int i = 0; i < 16; ++i) fv[i] = (i & bm) ? h1 : h0;
#pragma unroll
        for (int q = 0; q < 8; ++q) W[q] = W[q] * fv;
      }
    }

    // ---- carry hidden state across chunk boundary (mask = traj[:,7] = col 7)
    v8f d = {};
#pragma unroll
    for (int q = 0; q < 8; ++q)
      d = __builtin_amdgcn_wmma_f32_16x16x32_f16(false, W[q], false, Bf[q],
                                                 (short)0, d, false, false);
    float s1f[8], stf[8];
#pragma unroll
    for (int r = 0; r < 8; ++r) {
      s1f[r] = d[r];
      stf[r] = __shfl(d[r], srcTot, 32);
    }
    __syncthreads();
    if (bl == 7) {
#pragma unroll
      for (int r = 0; r < 8; ++r) {
        const int   m  = r + 8*laneHi;
        const float S1 = s1f[r];
        const float S0 = stf[r] - S1;
        const float N1 = cst[m][4]*S1 + cst[m][5]*S0;   // pF1*S1 + pL1*S0
        const float N0 = cst[m][6]*S1 + cst[m][7]*S0;   // pF0*S1 + pL0*S0
        const float iv = __frcp_rn(N1 + N0);            // renormalize (exact:
        nbuf[m][0] = N1 * iv;                           // outputs are ratios)
        nbuf[m][1] = N0 * iv;
      }
    }
    __syncthreads();
  }
}

extern "C" void kernel_launch(void* const* d_in, const int* in_sizes, int n_in,
                              void* d_out, int out_size, void* d_ws, size_t ws_size,
                              hipStream_t stream) {
  const float* corr = (const float*)d_in[0];
  const int*   kc   = (const int*)d_in[1];
  const float* lt   = (const float*)d_in[2];
  float*       out  = (float*)d_out;
  const int Bsz = in_sizes[1];                 // 4096 rows, 16 per wave
  fastbkt_wmma_kernel<<<dim3(Bsz / 16), dim3(32), 0, stream>>>(corr, kc, lt, out);
}
